// AttentionClassifier_13709535609431
// MI455X (gfx1250) — matrix operand order repaired and checked by reading, not compiled, not run on hardware
//
#include <hip/hip_runtime.h>
#include <hip/hip_bf16.h>

// ---------------------------------------------------------------------------
// Problem constants (B,J,D from the reference)
// ---------------------------------------------------------------------------
#define BB   64
#define JJ   512
#define DD   1024
#define MM   (BB * JJ)          // 32768 token rows

typedef __attribute__((ext_vector_type(16))) __bf16        v16bf;
typedef __attribute__((ext_vector_type(8)))  float         v8f;
typedef __attribute__((ext_vector_type(8)))  unsigned int  u32x8;

// round-to-nearest-even f32 -> bf16, packed pair (lo = first/even K)
__device__ __forceinline__ unsigned int pk_bf16(float lo, float hi) {
    unsigned int a = __float_as_uint(lo);
    a = a + 0x7FFFu + ((a >> 16) & 1u);
    unsigned int b = __float_as_uint(hi);
    b = b + 0x7FFFu + ((b >> 16) & 1u);
    return (a >> 16) | (b & 0xFFFF0000u);
}

// ---------------------------------------------------------------------------
// Kernel 1: W1 [K=1024][N=1024] f32  ->  W1T [N][K] bf16 (contiguous-K rows),
// plus zero the per-batch sums.  2 MB result lives in L2 for the whole GEMM.
// ---------------------------------------------------------------------------
__global__ __launch_bounds__(256)
void k_w1_to_bf16t(const float* __restrict__ W1,
                   unsigned short* __restrict__ w1t,
                   float* __restrict__ sumws) {
    __shared__ float tile[32][33];
    const int tx = threadIdx.x & 31;        // 32 x 8 thread tile
    const int ty = threadIdx.x >> 5;
    const int x0 = blockIdx.x * 32;         // N (column of W1)
    const int y0 = blockIdx.y * 32;         // K (row of W1)

    #pragma unroll
    for (int k = 0; k < 32; k += 8)
        tile[ty + k][tx] = W1[(size_t)(y0 + ty + k) * DD + (x0 + tx)];
    __syncthreads();
    #pragma unroll
    for (int k = 0; k < 32; k += 8) {
        float f = tile[tx][ty + k];         // = W1[y0+tx][x0+ty+k]
        unsigned int u = __float_as_uint(f);
        u = u + 0x7FFFu + ((u >> 16) & 1u);
        w1t[(size_t)(x0 + ty + k) * DD + (y0 + tx)] = (unsigned short)(u >> 16);
    }
    if (blockIdx.x == 0 && blockIdx.y == 0 && threadIdx.x < BB)
        sumws[threadIdx.x] = 0.0f;
}

// ---------------------------------------------------------------------------
// Kernel 2: fused  h = tanh(x@W1+b1);  val = sigmoid(h.W2+b2)*mask
//           -> valws[M], atomicAdd per-batch sums.
// 16 waves / block, wave (r,c):  rows [64*blk + 16r, +16),  cols [256c, +256).
// h never touches memory: it stays in the 16 f32 accumulator tiles per wave.
// ---------------------------------------------------------------------------
__global__ __launch_bounds__(512)
void k_fused_score(const float* __restrict__ x,
                   const float* __restrict__ maskp,
                   const unsigned short* __restrict__ w1t,
                   const float* __restrict__ b1,
                   const float* __restrict__ W2,
                   const float* __restrict__ b2,
                   float* __restrict__ valws,
                   float* __restrict__ sumws) {
    __shared__ float sval[64 * 4];

    const int tid  = threadIdx.x;
    const int wid  = tid >> 5;
    const int lane = tid & 31;
    const int r    = wid >> 2;              // 0..3  row-tile of 16
    const int c    = wid & 3;               // 0..3  256-col slab
    const int half = lane >> 4;             // lane group (ISA K-half select)
    const int lm   = lane & 15;

    const size_t row0  = (size_t)blockIdx.x * 64 + (size_t)r * 16;
    const int    ncol0 = c * 256;

    const v8f vzero = {0.f, 0.f, 0.f, 0.f, 0.f, 0.f, 0.f, 0.f};
    v8f acc[16];
    #pragma unroll
    for (int t = 0; t < 16; ++t) acc[t] = vzero;

    // A-fragment source: this lane holds row (row0+lm); ISA 16-bit A layout:
    //   v0..3 : K = kc + 8*half + {0..7}   (pairs per VGPR)
    //   v4..7 : K = kc + 16 + 8*half + {0..7}
    const float* __restrict__ xrow = x + (row0 + (size_t)lm) * DD;

    for (int kc = 0; kc < DD; kc += 32) {
        const float4 p0 = *(const float4*)(xrow + kc + half * 8);
        const float4 p1 = *(const float4*)(xrow + kc + half * 8 + 4);
        const float4 p2 = *(const float4*)(xrow + kc + 16 + half * 8);
        const float4 p3 = *(const float4*)(xrow + kc + 16 + half * 8 + 4);
        u32x8 au;
        au[0] = pk_bf16(p0.x, p0.y);  au[1] = pk_bf16(p0.z, p0.w);
        au[2] = pk_bf16(p1.x, p1.y);  au[3] = pk_bf16(p1.z, p1.w);
        au[4] = pk_bf16(p2.x, p2.y);  au[5] = pk_bf16(p2.z, p2.w);
        au[6] = pk_bf16(p3.x, p3.y);  au[7] = pk_bf16(p3.z, p3.w);
        const v16bf afrag = __builtin_bit_cast(v16bf, au);

        #pragma unroll
        for (int nt = 0; nt < 16; ++nt) {
            // B 32x16 bf16: lane = column n, K = 8*half+{0..7} then 16+8*half+{0..7},
            // two K per VGPR -> two b128 loads from W1T[n][kc+8*half..] and [kc+16+8*half..]
            const unsigned short* bp =
                w1t + (size_t)(ncol0 + nt * 16 + lm) * DD + kc + half * 8;
            const uint4 q0 = *(const uint4*)(bp);
            const uint4 q1 = *(const uint4*)(bp + 16);
            u32x8 bu;
            bu[0] = q0.x; bu[1] = q0.y; bu[2] = q0.z; bu[3] = q0.w;
            bu[4] = q1.x; bu[5] = q1.y; bu[6] = q1.z; bu[7] = q1.w;
            const v16bf bfrag = __builtin_bit_cast(v16bf, bu);

            acc[nt] = __builtin_amdgcn_wmma_f32_16x16x32_bf16(
                false, afrag, false, bfrag, (short)0, acc[nt], false, false);
        }
    }

    // Epilogue: bias + tanh, dot with W2 over this wave's 256 columns.
    // C layout: (vgpr i, lane) -> n = ncol0 + nt*16 + lm,  m = row0 + i + 8*half
    float rp[8];
    #pragma unroll
    for (int i = 0; i < 8; ++i) rp[i] = 0.f;

    #pragma unroll
    for (int nt = 0; nt < 16; ++nt) {
        const int n  = ncol0 + nt * 16 + lm;
        const float bias = b1[n];
        const float wn   = W2[n];
        #pragma unroll
        for (int i = 0; i < 8; ++i) {
            const float hv = tanhf(acc[nt][i] + bias);
            rp[i] += hv * wn;
        }
    }
    // reduce across the 16 lanes of each half (xor < 16 stays inside the group)
    #pragma unroll
    for (int off = 1; off < 16; off <<= 1) {
        #pragma unroll
        for (int i = 0; i < 8; ++i) rp[i] += __shfl_xor(rp[i], off, 32);
    }
    if (lm == 0) {
        #pragma unroll
        for (int i = 0; i < 8; ++i)
            sval[(r * 16 + half * 8 + i) * 4 + c] = rp[i];
    }
    __syncthreads();

    if (tid < 64) {
        float v = sval[tid * 4 + 0] + sval[tid * 4 + 1] +
                  sval[tid * 4 + 2] + sval[tid * 4 + 3] + b2[0];
        float s = 1.0f / (1.0f + expf(-v));                   // sigmoid
        const size_t gm = (size_t)blockIdx.x * 64 + tid;
        const int bb = (int)(gm >> 9);
        s *= maskp[gm];                                       // mask[b,j] flat
        valws[gm] = s;
        atomicAdd(&sumws[bb], s);
    }
}

// ---------------------------------------------------------------------------
// Kernel 3: per batch — normalize att weights (to output), weighted pooling
// of x over J, then the tiny W3 projection.  One block per batch.
// ---------------------------------------------------------------------------
__global__ __launch_bounds__(256)
void k_finalize(const float* __restrict__ x,
                const float* __restrict__ W3,
                const float* __restrict__ b3,
                const float* __restrict__ valws,
                const float* __restrict__ sumws,
                float* __restrict__ outv,     // [64,3]
                float* __restrict__ outatt) { // [64,512]
    __shared__ float satt[JJ];
    __shared__ float sred[3 * 256];

    const int b   = blockIdx.x;
    const int tid = threadIdx.x;
    const float inv = 1.0f / sumws[b];

    for (int j = tid; j < JJ; j += 256) {
        const float a = valws[(size_t)b * JJ + j] * inv;
        satt[j] = a;
        outatt[(size_t)b * JJ + j] = a;
    }
    __syncthreads();

    float pooled[4] = {0.f, 0.f, 0.f, 0.f};
    const float* __restrict__ xb = x + (size_t)b * JJ * DD;
    for (int j = 0; j < JJ; ++j) {
        const float a = satt[j];
        const float* __restrict__ xr = xb + (size_t)j * DD;
        #pragma unroll
        for (int q = 0; q < 4; ++q)
            pooled[q] += xr[tid + q * 256] * a;
    }

    float po[3] = {0.f, 0.f, 0.f};
    #pragma unroll
    for (int q = 0; q < 4; ++q) {
        const int d = tid + q * 256;
        #pragma unroll
        for (int o = 0; o < 3; ++o) po[o] += pooled[q] * W3[d * 3 + o];
    }
    #pragma unroll
    for (int o = 0; o < 3; ++o) sred[o * 256 + tid] = po[o];
    __syncthreads();
    for (int s = 128; s > 0; s >>= 1) {
        if (tid < s) {
            #pragma unroll
            for (int o = 0; o < 3; ++o)
                sred[o * 256 + tid] += sred[o * 256 + tid + s];
        }
        __syncthreads();
    }
    if (tid < 3) outv[b * 3 + tid] = sred[tid * 256] + b3[tid];
}

// ---------------------------------------------------------------------------
extern "C" void kernel_launch(void* const* d_in, const int* in_sizes, int n_in,
                              void* d_out, int out_size, void* d_ws, size_t ws_size,
                              hipStream_t stream) {
    const float* x     = (const float*)d_in[0];   // [64,512,1024]
    const float* maskp = (const float*)d_in[1];   // [64,512]
    const float* W1    = (const float*)d_in[2];   // [1024,1024]
    const float* b1    = (const float*)d_in[3];   // [1024]
    const float* W2    = (const float*)d_in[4];   // [1024,1]
    const float* b2    = (const float*)d_in[5];   // [1]
    const float* W3    = (const float*)d_in[6];   // [1024,3]
    const float* b3    = (const float*)d_in[7];   // [3]

    float* outv   = (float*)d_out;                // [64,3]  first 192 floats
    float* outatt = (float*)d_out + BB * 3;       // [64,512,1]

    unsigned short* w1t = (unsigned short*)d_ws;                 // 2 MB bf16 W1^T
    float* valws = (float*)((char*)d_ws + (size_t)DD * DD * 2);  // 32768 f32
    float* sumws = valws + MM;                                   // 64 f32

    // 1) W1 -> bf16 transposed, zero batch sums
    k_w1_to_bf16t<<<dim3(DD / 32, DD / 32), 256, 0, stream>>>(W1, w1t, sumws);
    // 2) fused GEMM (WMMA bf16) + tanh + W2 dot + sigmoid + mask + batch sums
    k_fused_score<<<MM / 64, 512, 0, stream>>>(x, maskp, w1t, b1, W2, b2,
                                               valws, sumws);
    // 3) per-batch normalize + pooled + W3 head
    k_finalize<<<BB, 256, 0, stream>>>(x, W3, b3, valws, sumws, outv, outatt);
}
